// Projector_31988916420622
// MI455X (gfx1250) — matrix operand order repaired and checked by reading, not compiled, not run on hardware
//
#include <hip/hip_runtime.h>

// ---------------------------------------------------------------------------
// Projector (cross-attention) for MI455X / gfx1250, wave32 + WMMA f16->f32.
// N=4, CT=512, C=512, L=256, S=H*W=4096, HEAD=16, d=32, GROUPS=16 (== HEAD).
//
// L-permutation trick: sigma(l) = (l%16)*16 + l/16 (self-inverse) is applied
// to the L dimension of the V workspace AND the softmax-coefficient store.
// P@V^T is invariant under a consistent permutation of the contraction dim,
// and softmax is order-invariant, so results are unchanged -- but the P store
// becomes one contiguous 32B v16h store per row instead of a 16x b16 scatter.
// ---------------------------------------------------------------------------

typedef __attribute__((ext_vector_type(16))) _Float16 v16h;
typedef __attribute__((ext_vector_type(8)))  _Float16 v8h;
typedef __attribute__((ext_vector_type(8)))  float    v8f;
typedef int b128_t __attribute__((vector_size(16)));   // matches builtin param

namespace cfg {
constexpr int N_     = 4;
constexpr int CT_    = 512;
constexpr int C_     = 512;
constexpr int L_     = 256;
constexpr int S_     = 4096;   // H*W
constexpr int HEAD_  = 16;
constexpr int D_     = 32;     // C/HEAD (== C/GROUPS)
constexpr int STILE_ = 128;    // query rows per workgroup
}

static __device__ __forceinline__ v8f v8f_zero() {
  v8f z;
#pragma unroll
  for (int i = 0; i < 8; ++i) z[i] = 0.0f;
  return z;
}

// sigma: self-inverse permutation of [0,256): transpose of the 16x16 l-grid
static __device__ __forceinline__ int lperm(int l) {
  return ((l & 15) << 4) | (l >> 4);
}

// --- CDNA5 async global->LDS copy (ASYNCcnt-tracked), with safe fallback ---
#if __has_builtin(__builtin_amdgcn_global_load_async_to_lds_b128)
#define HAVE_ASYNC_LDS 1
static __device__ __forceinline__ void async_copy16(void* lds, const void* g) {
  __builtin_amdgcn_global_load_async_to_lds_b128(
      (__attribute__((address_space(1))) b128_t*)(uintptr_t)g,
      (__attribute__((address_space(3))) b128_t*)(unsigned)(uintptr_t)lds,
      0, 0);
}
#else
#define HAVE_ASYNC_LDS 0
static __device__ __forceinline__ void async_copy16(void* lds, const void* g) {
  *(float4*)lds = *(const float4*)g;   // synchronous fallback
}
#endif

static __device__ __forceinline__ void wait_async0() {
#if HAVE_ASYNC_LDS
#if __has_builtin(__builtin_amdgcn_s_wait_asynccnt)
  __builtin_amdgcn_s_wait_asynccnt(0);
#else
  asm volatile("s_wait_asynccnt 0x0" ::: "memory");
#endif
#endif
}

// ===========================================================================
// Kernel 1: K/V projections (1x1 conv1d == GEMM W[C,CT] x token[n][CT,L]).
// Templated on PROJ (0 = V, 1 = K) so the store path is branch-free.
//   Vh[(n*C + c)*L + sigma(l)]               (d-major rows, permuted l)
//   Kh[((n*HEAD + hd)*L + l)*D + k]          (l-major rows, d contiguous)
// Grid: 64 blocks of 256 threads per instantiation.
//   blockIdx.x bits: [1:0]=l-strip(64), [3:2]=c-block(128), [5:4]=n
// Each wave owns one 16-row c tile x 4 l tiles; K-loop over CT in 32-chunks
// with the token chunk staged transposed ([l][ct], f16) in LDS.
// ===========================================================================
template <int PROJ>
__global__ __launch_bounds__(256) void kv_proj_kernel(
    const float* __restrict__ token,
    const float* __restrict__ Wgt, const float* __restrict__ bias,
    _Float16* __restrict__ dst) {
  using namespace cfg;
  __shared__ __align__(32) _Float16 tok[64][32];   // [l][ct] chunk, 4 KB

  const int bx    = blockIdx.x;
  const int strip = bx & 3;          // l-strip of 64
  const int cblk  = (bx >> 2) & 3;   // c-block of 128
  const int n     = bx >> 4;
  const int l0    = strip * 64;
  const int c0    = cblk * 128;

  const int lane = threadIdx.x & 31;
  const int wv   = threadIdx.x >> 5;
  const int m    = lane & 15;            // tile column / A row
  const int hi   = (lane & 16) ? 1 : 0;  // upper half-wave flag
  const int ctile_c = c0 + wv * 16;

  v8f acc[4];
#pragma unroll
  for (int t = 0; t < 4; ++t) acc[t] = v8f_zero();

  for (int ct0 = 0; ct0 < CT_; ct0 += 32) {
    __syncthreads();   // protect previous iteration's B reads
    // --- stage token chunk, transposed to [l][ct], f32 -> f16 ---
    {
      const int t   = threadIdx.x;
      const int ctl = t >> 3;            // 0..31
      const int lb  = (t & 7) * 8;       // 0..56
      const float* src = token + ((size_t)n * CT_ + ct0 + ctl) * L_ + l0 + lb;
#pragma unroll
      for (int j = 0; j < 8; ++j) tok[lb + j][ctl] = (_Float16)src[j];
    }
    __syncthreads();

    // --- A operand: 16(c) x 32(ct) strip of W, f32 -> f16, A layout ---
    v16h a;
    {
      const float* wr = Wgt + (size_t)(ctile_c + m) * CT_ + ct0;
      const int k1 = hi ? 16 : 0;
#pragma unroll
      for (int h = 0; h < 8; ++h) a[h]     = (_Float16)wr[k1 + h];
#pragma unroll
      for (int h = 0; h < 8; ++h) a[8 + h] = (_Float16)wr[8 + k1 + h];
    }

    // --- 4 l tiles: B from LDS, accumulate ---
#pragma unroll
    for (int lt = 0; lt < 4; ++lt) {
      const int l = lt * 16 + m;
      const v16h b = *(const v16h*)&tok[l][hi ? 16 : 0];
      acc[lt] = __builtin_amdgcn_wmma_f32_16x16x32_f16(
          false, a, false, b, (short)0, acc[lt], false, false);
    }
  }

  // --- bias + store (branch-free: PROJ is a template constant) ---
#pragma unroll
  for (int r = 0; r < 8; ++r) {
    const int c = ctile_c + r + (hi ? 8 : 0);
    const float bsv = bias[c];
#pragma unroll
    for (int lt = 0; lt < 4; ++lt) {
      const int l = l0 + lt * 16 + m;
      const float v = acc[lt][r] + bsv;
      if (PROJ == 0) {
        dst[((size_t)n * C_ + c) * L_ + lperm(l)] = (_Float16)v;
      } else {
        dst[(((size_t)n * HEAD_ + (c >> 5)) * L_ + l) * D_ + (c & 31)] = (_Float16)v;
      }
    }
  }
}

// ===========================================================================
// Kernel 2: fused Q-projection + attention + residual.
// Grid: (S/128, HEAD, N), 256 threads (8 waves), dynamic LDS 124 KB:
//   sF  [32 ch][128 s] f32 (16 KB) : feature tile (Q-proj input + residual acc)
//   sWq [32 o][32 i]   f32 ( 4 KB)
//   sQ  [128 s][32 d]  f16 ( 8 KB)
//   sP  [128 s][256 j] f16 (64 KB) : softmax coefs, j = sigma(l)
//   sK  [256 l][32 d]  f16 (16 KB) : async-staged K tile
//   sV  [32 d][256 j]  f16 (16 KB) : async-staged V tile (j = sigma(l))
// Wave w owns query rows [w*16, w*16+16).
// ===========================================================================
__global__ __launch_bounds__(256) void attn_kernel(
    const float* __restrict__ feature,
    const float* __restrict__ Wq, const float* __restrict__ bq,
    const _Float16* __restrict__ Vh, const _Float16* __restrict__ Kh,
    float* __restrict__ out) {
  using namespace cfg;
  extern __shared__ __align__(32) char smem_raw[];
  float*    sF  = (float*)smem_raw;                     // 32*128 f32
  float*    sWq = (float*)(smem_raw + 16 * 1024);       // 32*32  f32
  _Float16* sQ  = (_Float16*)(smem_raw + 20 * 1024);    // 128*32 f16
  _Float16* sP  = (_Float16*)(smem_raw + 28 * 1024);    // 128*256 f16
  _Float16* sK  = (_Float16*)(smem_raw + 92 * 1024);    // 256*32 f16
  _Float16* sV  = (_Float16*)(smem_raw + 108 * 1024);   // 32*256 f16

  const int n    = blockIdx.z;
  const int head = blockIdx.y;
  const int s0   = blockIdx.x * STILE_;

  const int lane = threadIdx.x & 31;
  const int wv   = threadIdx.x >> 5;
  const int m    = lane & 15;
  const int hi   = (lane & 16) ? 1 : 0;

  // ---- kick off async K/V tile staging (16 KB each, linear copies) ----
  {
    const _Float16* Kg = Kh + ((size_t)n * HEAD_ + head) * L_ * D_;
    const _Float16* Vg = Vh + ((size_t)n * HEAD_ + head) * D_ * L_;
    const int t = threadIdx.x;
#pragma unroll
    for (int i = 0; i < 4; ++i) {
      const int e = t + i * 256;           // 16-byte chunk id, 1024 total
      async_copy16(sK + e * 8, Kg + e * 8);
      async_copy16(sV + e * 8, Vg + e * 8);
    }
  }

  // ---- load feature tile (coalesced) + Wq[head] ----
  {
    const int t  = threadIdx.x;
    const int ch = t >> 3;
    const int sb = (t & 7) * 16;
    const float* src = feature + ((size_t)n * C_ + head * D_ + ch) * S_ + s0 + sb;
    float* dst = sF + ch * 128 + sb;
#pragma unroll
    for (int j = 0; j < 16; j += 4)
      *(float4*)(dst + j) = *(const float4*)(src + j);
    const float* wsrc = Wq + (size_t)head * D_ * D_;
#pragma unroll
    for (int j = 0; j < 4; ++j) sWq[t * 4 + j] = wsrc[t * 4 + j];
  }
  __syncthreads();

  // ---- Phase 1: Q projection.  Q^T[s,o] = F^T[s,i] x Wq^T[i,o] + bq ----
  {
    v16h a;   // A = F^T tile: A[m,k] = sF[k][wv*16+m]
    {
      const int scol = wv * 16 + m;
      const int kb = hi ? 16 : 0;
#pragma unroll
      for (int h = 0; h < 8; ++h) a[h]     = (_Float16)sF[(kb + h) * 128 + scol];
#pragma unroll
      for (int h = 0; h < 8; ++h) a[8 + h] = (_Float16)sF[(8 + kb + h) * 128 + scol];
    }
#pragma unroll
    for (int ot = 0; ot < 2; ++ot) {
      v16h b;   // B[k,n] = Wq[o = ot*16+n][i = k]
      const int o = ot * 16 + m;
      const int kk = hi ? 16 : 0;
#pragma unroll
      for (int h = 0; h < 16; ++h) b[h] = (_Float16)sWq[o * 32 + kk + h];
      v8f c = v8f_zero();
      c = __builtin_amdgcn_wmma_f32_16x16x32_f16(
          false, a, false, b, (short)0, c, false, false);
      const float bb = bq[(size_t)head * D_ + o];
#pragma unroll
      for (int r = 0; r < 8; ++r) {
        const int srow = wv * 16 + r + (hi ? 8 : 0);
        sQ[srow * 32 + o] = (_Float16)(c[r] + bb);
      }
    }
  }
  wait_async0();      // K/V tiles resident in LDS after the barrier below
  __syncthreads();

  // ---- Phase 2: scores = Q K / sqrt(d), softmax over L=256 ----
  {
    const float scale = 0.17677669529663687f;   // 1/sqrt(32)
    v16h a;   // A = Q strip rows [wv*16, +16)
    {
      const _Float16* qrow = sQ + (wv * 16 + m) * 32;
      const int k1 = hi ? 16 : 0;
#pragma unroll
      for (int h = 0; h < 8; ++h) a[h]     = qrow[k1 + h];
#pragma unroll
      for (int h = 0; h < 8; ++h) a[8 + h] = qrow[8 + k1 + h];
    }
    v8f acc[16];
#pragma unroll
    for (int lt = 0; lt < 16; ++lt) {
      // B: K[d, l-tile]: lane holds 16 contiguous f16 of row l, d-chunk hi*16
      const v16h b = *(const v16h*)(sK + (size_t)(lt * 16 + m) * D_ + (hi ? 16 : 0));
      acc[lt] = __builtin_amdgcn_wmma_f32_16x16x32_f16(
          false, a, false, b, (short)0, v8f_zero(), false, false);
    }
    // softmax: row (wv*16 + r + 8*hi) lives in one 16-lane half-wave.
    // Lane m holds columns l = lt*16 + m; their sigma-indices are the 16
    // CONTIGUOUS slots [m*16, m*16+16) -> single v16h store per row.
#pragma unroll
    for (int r = 0; r < 8; ++r) {
      float mx = -3.0e38f;
#pragma unroll
      for (int lt = 0; lt < 16; ++lt) mx = fmaxf(mx, acc[lt][r] * scale);
#pragma unroll
      for (int off = 1; off < 16; off <<= 1) mx = fmaxf(mx, __shfl_xor(mx, off, 32));
      float e[16], sum = 0.0f;
#pragma unroll
      for (int lt = 0; lt < 16; ++lt) {
        e[lt] = __expf(acc[lt][r] * scale - mx);
        sum += e[lt];
      }
#pragma unroll
      for (int off = 1; off < 16; off <<= 1) sum += __shfl_xor(sum, off, 32);
      const float rinv = 1.0f / sum;
      v16h pv;
#pragma unroll
      for (int lt = 0; lt < 16; ++lt) pv[lt] = (_Float16)(e[lt] * rinv);
      _Float16* prow = sP + (size_t)(wv * 16 + r + (hi ? 8 : 0)) * 256;
      *(v16h*)(prow + m * 16) = pv;
    }
  }
  __syncthreads();

  // ---- Phase 3: out^T[s, o] = P[s, j] x Vperm^T[j, o]; residual into sF ----
  {
    v8f oacc[2];
    oacc[0] = v8f_zero();
    oacc[1] = v8f_zero();
#pragma unroll
    for (int lc = 0; lc < 8; ++lc) {
      v16h a;   // A = P chunk [16 s, 32 j]
      {
        const _Float16* prow = sP + (size_t)(wv * 16 + m) * 256 + lc * 32;
        const int k1 = hi ? 16 : 0;
#pragma unroll
        for (int h = 0; h < 8; ++h) a[h]     = prow[k1 + h];
#pragma unroll
        for (int h = 0; h < 8; ++h) a[8 + h] = prow[8 + k1 + h];
      }
#pragma unroll
      for (int ot = 0; ot < 2; ++ot) {
        // B = Vperm^T chunk: lane holds sV[o][lc*32 + hi*16 .. +16) contiguous
        const v16h b = *(const v16h*)(sV + (size_t)(ot * 16 + m) * L_ + lc * 32 + (hi ? 16 : 0));
        oacc[ot] = __builtin_amdgcn_wmma_f32_16x16x32_f16(
            false, a, false, b, (short)0, oacc[ot], false, false);
      }
    }
#pragma unroll
    for (int ot = 0; ot < 2; ++ot) {
      const int o = ot * 16 + m;
#pragma unroll
      for (int r = 0; r < 8; ++r) {
        const int srow = wv * 16 + r + (hi ? 8 : 0);   // wave-private rows
        sF[o * 128 + srow] += oacc[ot][r];
      }
    }
  }
  __syncthreads();

  // ---- store out = feature + proj (coalesced) ----
  {
    const int t  = threadIdx.x;
    const int ch = t >> 3;
    const int sb = (t & 7) * 16;
    float* dstg = out + ((size_t)n * C_ + head * D_ + ch) * S_ + s0 + sb;
    const float* srcl = sF + ch * 128 + sb;
#pragma unroll
    for (int j = 0; j < 16; j += 4)
      *(float4*)(dstg + j) = *(const float4*)(srcl + j);
  }
}

// ===========================================================================
extern "C" void kernel_launch(void* const* d_in, const int* in_sizes, int n_in,
                              void* d_out, int out_size, void* d_ws, size_t ws_size,
                              hipStream_t stream) {
  using namespace cfg;
  (void)in_sizes; (void)n_in; (void)out_size; (void)ws_size;
  const float* feature = (const float*)d_in[0];
  const float* token   = (const float*)d_in[1];
  const float* Wv      = (const float*)d_in[2];
  const float* bv      = (const float*)d_in[3];
  const float* Wk      = (const float*)d_in[4];
  const float* bk      = (const float*)d_in[5];
  const float* Wq      = (const float*)d_in[6];
  const float* bq      = (const float*)d_in[7];
  float* out = (float*)d_out;

  _Float16* Vh = (_Float16*)d_ws;                      // N*C*L f16 = 1 MB
  _Float16* Kh = Vh + (size_t)N_ * C_ * L_;            // N*C*L f16 = 1 MB

  kv_proj_kernel<0><<<64, 256, 0, stream>>>(token, Wv, bv, Vh);
  kv_proj_kernel<1><<<64, 256, 0, stream>>>(token, Wk, bk, Kh);

  const size_t smem = 124 * 1024;   // sF+sWq+sQ+sP+sK+sV
  attn_kernel<<<dim3(S_ / STILE_, HEAD_, N_), 256, smem, stream>>>(
      feature, Wq, bq, Vh, Kh, out);
}
